// GNN_Sage_10153302688447
// MI455X (gfx1250) — matrix operand order repaired and checked by reading, not compiled, not run on hardware
//
#include <hip/hip_runtime.h>
#include <hip/hip_bf16.h>
#include <math.h>

// ---------------- constants (from reference) ----------------
#define N_NODES  100000
#define N_EDGES  800000
#define N_GRAPHS 64
#define D_IN     128
#define D_H      256
#define D_MID    64
#define D_OUT    3

// ---------------- CDNA5 WMMA types ----------------
typedef __attribute__((ext_vector_type(16))) __bf16 v16bf;
typedef __attribute__((ext_vector_type(8)))  float  v8f;

union FragAB { v16bf v; uint4 u[2]; };

static __device__ __forceinline__ unsigned short f2bf(float f) {
    unsigned u = __float_as_uint(f);
    // round-to-nearest-even bf16
    unsigned r = (u + 0x7FFFu + ((u >> 16) & 1u)) >> 16;
    return (unsigned short)r;
}

// ---------------- weight prep: Wt[n][k] = concat(Wl;Wr)[k][n] in bf16 ----------------
__global__ void k_prep_w(const float* __restrict__ Wl, const float* __restrict__ Wr,
                         int d, int N, unsigned short* __restrict__ Wt) {
    int t = blockIdx.x * blockDim.x + threadIdx.x;
    int total = N * 2 * d;
    if (t >= total) return;
    int n = t / (2 * d);
    int k = t - n * (2 * d);
    float v = (k < d) ? Wl[(size_t)k * N + n] : Wr[(size_t)(k - d) * N + n];
    Wt[(size_t)n * (2 * d) + k] = f2bf(v);
}

// ---------------- degree ----------------
__global__ void k_degree(const int* __restrict__ dst, float* __restrict__ deg, int E) {
    int t = blockIdx.x * blockDim.x + threadIdx.x;
    if (t >= E) return;
    atomicAdd(&deg[dst[t]], 1.0f);
}
__global__ void k_rdeg(const float* __restrict__ deg, float* __restrict__ rdeg, int M) {
    int t = blockIdx.x * blockDim.x + threadIdx.x;
    if (t >= M) return;
    rdeg[t] = 1.0f / fmaxf(deg[t], 1.0f);
}

// ---------------- edge aggregation: msum[dst] += h[src] (float4 chunks) ----------------
__global__ void k_aggregate(const float* __restrict__ h, const int* __restrict__ src,
                            const int* __restrict__ dst, float* __restrict__ msum,
                            int E, int dq /* = d/4 */) {
    int t = blockIdx.x * blockDim.x + threadIdx.x;
    int total = E * dq;
    if (t >= total) return;
    int e = t / dq;
    int c = t - e * dq;
    int s = src[e], d_ = dst[e];
    const float4 v = ((const float4*)(h + (size_t)s * (dq * 4)))[c];
    float* o = msum + (size_t)d_ * (dq * 4) + c * 4;
    atomicAdd(o + 0, v.x);
    atomicAdd(o + 1, v.y);
    atomicAdd(o + 2, v.z);
    atomicAdd(o + 3, v.w);
}

// ---------------- finalize: X2 = [mean | h] as bf16, row length 2d ----------------
__global__ void k_finalize(const float* __restrict__ msum, const float* __restrict__ h,
                           const float* __restrict__ rdeg, unsigned short* __restrict__ X2,
                           int M, int d) {
    int t = blockIdx.x * blockDim.x + threadIdx.x;
    if (t >= M * d) return;
    int m = t / d;
    int j = t - m * d;
    size_t o = (size_t)m * (2 * d);
    X2[o + j]     = f2bf(msum[t] * rdeg[m]);
    X2[o + d + j] = f2bf(h[t]);
}

// ---------------- WMMA GEMM with double-buffered LDS weights ----------------
// C[M x 256] = A[M x K](bf16) * Wt[256 x K]^T + bias.
// 10 waves / block (320 threads); each wave owns a 16(M) x 256(N) strip.
// Weight K-chunks are staged in LDS (2 x 32 KB ping-pong) and shared by all
// 10 waves; B fragments are register double-buffered so WMMAs wait on
// dscnt<=2 instead of 0 (LDS latency overlapped with the matrix pipe).
#define KC 64                      // K-chunk staged in LDS (256*KC*2B = 32 KB)
#define GEMM_WAVES 10
#define GEMM_THREADS (GEMM_WAVES * 32)

__global__ __launch_bounds__(GEMM_THREADS)
void k_gemm_wmma(const unsigned short* __restrict__ A, const unsigned short* __restrict__ Wt,
                 const float* __restrict__ bias, float* __restrict__ C,
                 int M, int K, int do_tanh) {
    __shared__ __align__(16) unsigned short Bs[2][256 * KC];

    const int lane = threadIdx.x & 31;
    const int waveInBlk = threadIdx.x >> 5;                     // 0..9
    const int m0 = (blockIdx.x * GEMM_WAVES + waveInBlk) << 4;  // grid sized exactly
    const int half = lane >> 4;                                 // 0 | 1
    const int l16  = lane & 15;

    const v8f vzero = {0.f, 0.f, 0.f, 0.f, 0.f, 0.f, 0.f, 0.f};
    v8f acc[16];
#pragma unroll
    for (int i = 0; i < 16; ++i) acc[i] = vzero;

    // A-frag (16x32 bf16): row = m0 + l16; 16 contiguous K at half*16
    const unsigned short* aBase = A + (size_t)(m0 + l16) * K + half * 16;

    // cooperative stage of Wt[:, kc:kc+KC] -> Bs[buf] (uint4 = 8 bf16 per load)
    auto stage = [&](int kc, int buf) {
        for (int i = threadIdx.x; i < (256 * KC) / 8; i += GEMM_THREADS) {
            int n = i >> 3;                 // KC/8 == 8 vectors per row
            int koff = (i & 7) << 3;
            *reinterpret_cast<uint4*>(&Bs[buf][n * KC + koff]) =
                *reinterpret_cast<const uint4*>(Wt + (size_t)n * K + kc + koff);
        }
    };

    stage(0, 0);
    __syncthreads();

    const int nCh = K / KC;
    for (int c = 0; c < nCh; ++c) {
        // prefetch next weight chunk into the other LDS buffer while computing
        if (c + 1 < nCh) stage((c + 1) * KC, (c + 1) & 1);

        // B-frag base in LDS: col = nt*16 + l16; 16 contiguous K at half*16
        const unsigned short* bls = Bs[c & 1] + l16 * KC + half * 16;
        const int kc = c * KC;

#pragma unroll
        for (int kk = 0; kk < KC; kk += 32) {
            FragAB a;
            a.u[0] = *reinterpret_cast<const uint4*>(aBase + kc + kk);
            a.u[1] = *reinterpret_cast<const uint4*>(aBase + kc + kk + 8);

            FragAB b[2];
            b[0].u[0] = *reinterpret_cast<const uint4*>(bls + kk);
            b[0].u[1] = *reinterpret_cast<const uint4*>(bls + kk + 8);
#pragma unroll
            for (int nt = 0; nt < 16; ++nt) {
                if (nt < 15) {   // load next B fragment into the alternate regs
                    const unsigned short* bp = bls + (nt + 1) * 16 * KC + kk;
                    b[(nt + 1) & 1].u[0] = *reinterpret_cast<const uint4*>(bp);
                    b[(nt + 1) & 1].u[1] = *reinterpret_cast<const uint4*>(bp + 8);
                }
                acc[nt] = __builtin_amdgcn_wmma_f32_16x16x32_bf16(
                    false, a.v, false, b[nt & 1].v, (short)0, acc[nt], false, false);
            }
        }
        __syncthreads();   // stage(c+1) done for everyone; chunk c consumed
    }

    const int mrowBase = m0 + half * 8;
#pragma unroll
    for (int nt = 0; nt < 16; ++nt) {
        const int n = nt * 16 + l16;
        const float bn = bias[n];
#pragma unroll
        for (int r = 0; r < 8; ++r) {
            float v = acc[nt][r] + bn;
            if (do_tanh) v = tanhf(v);
            C[(size_t)(mrowBase + r) * 256 + n] = v;
        }
    }
}

// ---------------- segment-max pooling ----------------
static __device__ __forceinline__ void atomicMaxF(float* a, float v) {
    if (v >= 0.f) atomicMax((int*)a, __float_as_int(v));
    else          atomicMin((unsigned int*)a, __float_as_uint(v));
}

__global__ void k_fill_u32(unsigned* __restrict__ p, unsigned v, int n) {
    int t = blockIdx.x * blockDim.x + threadIdx.x;
    if (t < n) p[t] = v;
}

#define POOL_CHUNK 16
__global__ void k_pool(const float* __restrict__ h, const int* __restrict__ batch,
                       float* __restrict__ pooled, int M) {
    int t = blockIdx.x * blockDim.x + threadIdx.x;
    int j = t & 255;
    int m0 = (t >> 8) * POOL_CHUNK;
    if (m0 >= M) return;
    int mEnd = (m0 + POOL_CHUNK < M) ? (m0 + POOL_CHUNK) : M;
    float best = -INFINITY;
    int curg = -1;
    for (int m = m0; m < mEnd; ++m) {
        int g = batch[m];
        if (g != curg) {
            if (curg >= 0) atomicMaxF(&pooled[(size_t)curg * 256 + j], best);
            curg = g;
            best = -INFINITY;
        }
        best = fmaxf(best, h[(size_t)m * 256 + j]);
    }
    if (curg >= 0) atomicMaxF(&pooled[(size_t)curg * 256 + j], best);
}

// ---------------- MLP head (tiny: 64x256 -> 64x64 -> 64x3) ----------------
__global__ void k_head_mid(const float* __restrict__ pooled, const float* __restrict__ W,
                           const float* __restrict__ b, float* __restrict__ mid) {
    int t = blockIdx.x * blockDim.x + threadIdx.x;
    if (t >= N_GRAPHS * D_MID) return;
    int g = t >> 6, j = t & 63;
    float s = b[j];
    for (int i = 0; i < D_H; ++i) {
        float p = pooled[(size_t)g * D_H + i];
        p = (p == -INFINITY) ? 0.f : p;     // empty segments -> 0
        s += p * W[(size_t)i * D_MID + j];
    }
    mid[t] = tanhf(s);
}

__global__ void k_head_out(const float* __restrict__ mid, const float* __restrict__ W,
                           const float* __restrict__ b, float* __restrict__ out) {
    int t = blockIdx.x * blockDim.x + threadIdx.x;
    if (t >= N_GRAPHS * D_OUT) return;
    int g = t / D_OUT, o = t - g * D_OUT;
    float s = b[o];
    for (int i = 0; i < D_MID; ++i)
        s += mid[(size_t)g * D_MID + i] * W[(size_t)i * D_OUT + o];
    out[t] = s;
}

// ---------------- launch ----------------
static inline int divUp(long long a, long long b) { return (int)((a + b - 1) / b); }

extern "C" void kernel_launch(void* const* d_in, const int* in_sizes, int n_in,
                              void* d_out, int out_size, void* d_ws, size_t ws_size,
                              hipStream_t stream) {
    const int M = N_NODES, E = N_EDGES;

    const float* x     = (const float*)d_in[0];
    const int*   ei    = (const int*)d_in[1];
    const int*   src   = ei;
    const int*   dst   = ei + E;
    const int*   batch = (const int*)d_in[2];
    const float *W1l = (const float*)d_in[3],  *b1l = (const float*)d_in[4],  *W1r = (const float*)d_in[5];
    const float *W2l = (const float*)d_in[6],  *b2l = (const float*)d_in[7],  *W2r = (const float*)d_in[8];
    const float *W3l = (const float*)d_in[9],  *b3l = (const float*)d_in[10], *W3r = (const float*)d_in[11];
    const float *Wlin1 = (const float*)d_in[12], *blin1 = (const float*)d_in[13];
    const float *Wlin2 = (const float*)d_in[14], *blin2 = (const float*)d_in[15];

    // ---- workspace carve (all 256B aligned) ----
    char* ws = (char*)d_ws;
    size_t off = 0;
    auto carve = [&](size_t bytes) -> void* {
        void* p = ws + off;
        off = (off + bytes + 255) & ~(size_t)255;
        return p;
    };
    float*          deg    = (float*)carve((size_t)M * 4);
    float*          rdeg   = (float*)carve((size_t)M * 4);
    float*          msum   = (float*)carve((size_t)M * 256 * 4);
    unsigned short* X2     = (unsigned short*)carve((size_t)M * 512 * 2);
    float*          hA     = (float*)carve((size_t)M * 256 * 4);
    float*          hB     = (float*)carve((size_t)M * 256 * 4);
    unsigned short* Wt1    = (unsigned short*)carve((size_t)256 * 256 * 2);
    unsigned short* Wt2    = (unsigned short*)carve((size_t)256 * 512 * 2);
    unsigned short* Wt3    = (unsigned short*)carve((size_t)256 * 512 * 2);
    float*          pooled = (float*)carve((size_t)N_GRAPHS * 256 * 4);
    float*          mid    = (float*)carve((size_t)N_GRAPHS * D_MID * 4);
    (void)ws_size; (void)in_sizes; (void)n_in; (void)out_size;

    // 6250 M-tiles, 10 waves per block -> exactly 625 blocks, all waves active
    const int gemmBlocks = (M / 16) / GEMM_WAVES;

    // ---- weight prep (small, L2-resident afterwards) ----
    k_prep_w<<<divUp(256 * 256, 256), 256, 0, stream>>>(W1l, W1r, 128, 256, Wt1);
    k_prep_w<<<divUp(256 * 512, 256), 256, 0, stream>>>(W2l, W2r, 256, 256, Wt2);
    k_prep_w<<<divUp(256 * 512, 256), 256, 0, stream>>>(W3l, W3r, 256, 256, Wt3);

    // ---- degrees (shared by all 3 layers) ----
    hipMemsetAsync(deg, 0, (size_t)M * 4, stream);
    k_degree<<<divUp(E, 256), 256, 0, stream>>>(dst, deg, E);
    k_rdeg<<<divUp(M, 256), 256, 0, stream>>>(deg, rdeg, M);

    // ---- layer 1: d_in=128, K=256 ----
    hipMemsetAsync(msum, 0, (size_t)M * 128 * 4, stream);
    k_aggregate<<<divUp((long long)E * 32, 256), 256, 0, stream>>>(x, src, dst, msum, E, 32);
    k_finalize<<<divUp((long long)M * 128, 256), 256, 0, stream>>>(msum, x, rdeg, X2, M, 128);
    k_gemm_wmma<<<gemmBlocks, GEMM_THREADS, 0, stream>>>(X2, Wt1, b1l, hA, M, 256, 1);

    // ---- layer 2: d=256, K=512 ----
    hipMemsetAsync(msum, 0, (size_t)M * 256 * 4, stream);
    k_aggregate<<<divUp((long long)E * 64, 256), 256, 0, stream>>>(hA, src, dst, msum, E, 64);
    k_finalize<<<divUp((long long)M * 256, 256), 256, 0, stream>>>(msum, hA, rdeg, X2, M, 256);
    k_gemm_wmma<<<gemmBlocks, GEMM_THREADS, 0, stream>>>(X2, Wt2, b2l, hB, M, 512, 1);

    // ---- layer 3: d=256, K=512, no tanh ----
    hipMemsetAsync(msum, 0, (size_t)M * 256 * 4, stream);
    k_aggregate<<<divUp((long long)E * 64, 256), 256, 0, stream>>>(hB, src, dst, msum, E, 64);
    k_finalize<<<divUp((long long)M * 256, 256), 256, 0, stream>>>(msum, hB, rdeg, X2, M, 256);
    k_gemm_wmma<<<gemmBlocks, GEMM_THREADS, 0, stream>>>(X2, Wt3, b3l, hA, M, 512, 0);

    // ---- global max pool ----
    k_fill_u32<<<divUp(N_GRAPHS * 256, 256), 256, 0, stream>>>((unsigned*)pooled, 0xFF800000u,
                                                               N_GRAPHS * 256);
    k_pool<<<divUp(M, POOL_CHUNK), 256, 0, stream>>>(hA, batch, pooled, M);

    // ---- head ----
    k_head_mid<<<divUp(N_GRAPHS * D_MID, 256), 256, 0, stream>>>(pooled, Wlin1, blin1, mid);
    k_head_out<<<1, 256, 0, stream>>>(mid, Wlin2, blin2, (float*)d_out);
}
